// DiscreteLSTMActor_33612414058818
// MI455X (gfx1250) — matrix operand order repaired and checked by reading, not compiled, not run on hardware
//
#include <hip/hip_runtime.h>
#include <hip/hip_bf16.h>

// ---------------- problem constants ----------------
#define TT    128
#define BB    256
#define TB    32768      // T*B
#define OBSN  1024
#define FEAT  512
#define AA    15
#define DD    528        // FEAT + A + 1
#define DPAD  544        // DD padded to multiple of 32 for WMMA K
#define G4    2112       // 4*DD

typedef __attribute__((ext_vector_type(16))) _Float16 v16h;
typedef __attribute__((ext_vector_type(8)))  float    v8f;

__device__ __forceinline__ v8f wmma16(v16h a, v16h b, v8f c) {
  // v_wmma_f32_16x16x32_f16  (neg_a, A, neg_b, B, c_mod, C, reuse_a, reuse_b)
  return __builtin_amdgcn_wmma_f32_16x16x32_f16(false, a, false, b, (short)0, c, false, false);
}

__device__ __forceinline__ float sigmoidf_(float x) { return 1.0f / (1.0f + expf(-x)); }

// Generic f16 GEMM micro-tile: one wave computes one 16-row M tile times NB
// consecutive 16-col N tiles, A fragment reused across the NB WMMAs.
// A: (M, lda) f16 row-major.  W: (N, ldk) f16 row-major (i.e. B^T storage).
// Fragment layouts used here (wave32, 16-bit operands; contiguous k, not ISA table order):
//   A: row = lane&15; elem e     -> K=k0+half*16+e   (16 contiguous f16/lane).
//   B: col = lane&15; elem e     -> K=k0+half*16+e   (16 contiguous f16/lane).
//   C/D: row = half*8+r (VGPR r), col = lane&15.
template<int NB>
__device__ __forceinline__ void gemm_nb(const _Float16* __restrict__ A, int lda,
                                        const _Float16* __restrict__ W, int ldk,
                                        int m0, int n0, int K, v8f* acc) {
  int lane = threadIdx.x & 31;
  int half = lane >> 4, r15 = lane & 15;
  const _Float16* arow = A + (long)(m0 + r15) * lda;
  for (int k0 = 0; k0 < K; k0 += 32) {
    const _Float16* ap = arow + k0 + half * 16;
    __builtin_prefetch((const void*)(ap + 64), 0, 0);  // A K-panel two steps ahead (speculative)
    v16h a;
#pragma unroll
    for (int e = 0; e < 8; ++e) a[e] = ap[e];
#pragma unroll
    for (int e = 0; e < 8; ++e) a[8 + e] = ap[8 + e];
#pragma unroll
    for (int i = 0; i < NB; ++i) {
      const _Float16* bp = W + (long)(n0 + i * 16 + r15) * ldk + k0 + half * 16;
      v16h b;
#pragma unroll
      for (int e = 0; e < 16; ++e) b[e] = bp[e];
      acc[i] = wmma16(a, b, acc[i]);
    }
  }
}

// ---------------- converts ----------------
__global__ void k_cvt(const float* __restrict__ src, _Float16* __restrict__ dst, long n) {
  long i = (long)blockIdx.x * blockDim.x + threadIdx.x;
  if (i < n) dst[i] = (_Float16)src[i];
}

__global__ void k_cvt_pad(const float* __restrict__ src, _Float16* __restrict__ dst,
                          int rows, int kin, int kpad) {
  long idx = (long)blockIdx.x * blockDim.x + threadIdx.x;
  long total = (long)rows * kpad;
  if (idx >= total) return;
  int r = (int)(idx / kpad), k = (int)(idx % kpad);
  dst[idx] = (_Float16)(k < kin ? src[(long)r * kin + k] : 0.0f);
}

__global__ void k_zero(float* __restrict__ p, long n) {
  long i = (long)blockIdx.x * blockDim.x + threadIdx.x;
  if (i < n) p[i] = 0.0f;
}

// ---------------- encoder: relu(obs @ enc_w^T) -> lstm_in cols 0..511 (f16) ----------------
__global__ void k_enc_gemm(const _Float16* __restrict__ obs_h, const _Float16* __restrict__ encw,
                           const float* __restrict__ encb, _Float16* __restrict__ lstm_in) {
  const int NG = FEAT / 64;  // 8 groups of 4 N-tiles
  int wave = threadIdx.x >> 5, lane = threadIdx.x & 31;
  int half = lane >> 4, r15 = lane & 15;
  int tile = blockIdx.x * 4 + wave;          // exact grid, EXEC all-1s
  int mt = tile / NG, ng = tile % NG;
  int m0 = mt * 16, n0 = ng * 64;
  v8f acc[4] = {};
  gemm_nb<4>(obs_h, OBSN, encw, OBSN, m0, n0, OBSN, acc);
#pragma unroll
  for (int i = 0; i < 4; ++i) {
    int col = n0 + i * 16 + r15;
    float bias = encb[col];
#pragma unroll
    for (int r = 0; r < 8; ++r) {
      float v = acc[i][r] + bias;
      v = v > 0.0f ? v : 0.0f;
      lstm_in[(long)(m0 + half * 8 + r) * DPAD + col] = (_Float16)v;
    }
  }
}

// cols 512 (clipped reward), 513..527 (one-hot), 528..543 (pad=0)
__global__ void k_fill_extras(const float* __restrict__ reward, const int* __restrict__ last_action,
                              _Float16* __restrict__ lstm_in) {
  int m = blockIdx.x * blockDim.x + threadIdx.x;
  if (m >= TB) return;
  float rw = reward[m];
  rw = fminf(1.0f, fmaxf(-1.0f, rw));
  int a = last_action[m];
  _Float16* row = lstm_in + (long)m * DPAD;
  row[FEAT] = (_Float16)rw;
#pragma unroll
  for (int j = 0; j < AA; ++j) row[FEAT + 1 + j] = (_Float16)(j == a ? 1.0f : 0.0f);
#pragma unroll
  for (int j = DD; j < DPAD; ++j) row[j] = (_Float16)0.0f;
}

// ---------------- gates_x = lstm_in @ w_ih0^T + b_ih0 + b_hh0 (f16 out) ----------------
__global__ void k_gatesx_gemm(const _Float16* __restrict__ lstm_in, const _Float16* __restrict__ wih0,
                              const float* __restrict__ bih, const float* __restrict__ bhh,
                              _Float16* __restrict__ gates_x) {
  const int NG = G4 / 64;  // 33
  int wave = threadIdx.x >> 5, lane = threadIdx.x & 31;
  int half = lane >> 4, r15 = lane & 15;
  int tile = blockIdx.x * 4 + wave;
  int mt = tile / NG, ng = tile % NG;
  int m0 = mt * 16, n0 = ng * 64;
  v8f acc[4] = {};
  gemm_nb<4>(lstm_in, DPAD, wih0, DPAD, m0, n0, DPAD, acc);
#pragma unroll
  for (int i = 0; i < 4; ++i) {
    int col = n0 + i * 16 + r15;
    float bias = bih[col] + bhh[col];
#pragma unroll
    for (int r = 0; r < 8; ++r)
      gates_x[(long)(m0 + half * 8 + r) * G4 + col] = (_Float16)(acc[i][r] + bias);
  }
}

// ---------------- per-step mask prep: hmf = notdone ? hf : 0 (full padded rows) ----------------
__global__ void k_prep_mask(const _Float16* __restrict__ h1f, const _Float16* __restrict__ h2f,
                            const int* __restrict__ term_t,
                            _Float16* __restrict__ h1mf, _Float16* __restrict__ h2mf) {
  int idx = blockIdx.x * blockDim.x + threadIdx.x;  // BB*DPAD = 139264
  if (idx >= BB * DPAD) return;
  int b = idx / DPAD;
  bool nd = term_t[b] == 0;
  h1mf[idx] = nd ? h1f[idx] : (_Float16)0.0f;
  h2mf[idx] = nd ? h2f[idx] : (_Float16)0.0f;
}

// ---------------- per-step cell0 GEMM: gates = gates_x[t] + h1m @ w_hh0^T ----------------
__global__ void k_cell0_gemm(const _Float16* __restrict__ h1mf, const _Float16* __restrict__ whh0,
                             const _Float16* __restrict__ gates_x_t, float* __restrict__ gates) {
  const int NG = G4 / 64;  // 33; Mtiles = 16 -> 528 waves -> 132 blocks exact
  int wave = threadIdx.x >> 5, lane = threadIdx.x & 31;
  int half = lane >> 4, r15 = lane & 15;
  int tile = blockIdx.x * 4 + wave;
  int mt = tile / NG, ng = tile % NG;
  int m0 = mt * 16, n0 = ng * 64;
  v8f acc[4] = {};
  gemm_nb<4>(h1mf, DPAD, whh0, DPAD, m0, n0, DPAD, acc);
#pragma unroll
  for (int i = 0; i < 4; ++i) {
    int col = n0 + i * 16 + r15;
#pragma unroll
    for (int r = 0; r < 8; ++r) {
      long row = m0 + half * 8 + r;
      gates[row * G4 + col] = acc[i][r] + (float)gates_x_t[row * G4 + col];
    }
  }
}

__global__ void k_cell0_elem(const float* __restrict__ gates, const int* __restrict__ term_t,
                             _Float16* __restrict__ h1f, float* __restrict__ c1) {
  int idx = blockIdx.x * blockDim.x + threadIdx.x;  // BB*DD = 135168
  if (idx >= BB * DD) return;
  int b = idx / DD, j = idx % DD;
  const float* g = gates + (long)b * G4;
  float gi = g[j], gf = g[DD + j], gg = g[2 * DD + j], go = g[3 * DD + j];
  float nd = term_t[b] ? 0.0f : 1.0f;
  float cp = c1[b * DPAD + j] * nd;
  float cn = sigmoidf_(gf) * cp + sigmoidf_(gi) * tanhf(gg);
  float hn = sigmoidf_(go) * tanhf(cn);
  c1[b * DPAD + j] = cn;
  h1f[b * DPAD + j] = (_Float16)hn;
}

// ---------------- per-step cell1 GEMM: gates = b1 + h1 @ w_ih1^T + h2m @ w_hh1^T ----------------
__global__ void k_cell1_gemm(const _Float16* __restrict__ h1f, const _Float16* __restrict__ h2mf,
                             const _Float16* __restrict__ wih1, const _Float16* __restrict__ whh1,
                             const float* __restrict__ bih1, const float* __restrict__ bhh1,
                             float* __restrict__ gates) {
  const int NG = G4 / 64;
  int wave = threadIdx.x >> 5, lane = threadIdx.x & 31;
  int half = lane >> 4, r15 = lane & 15;
  int tile = blockIdx.x * 4 + wave;
  int mt = tile / NG, ng = tile % NG;
  int m0 = mt * 16, n0 = ng * 64;
  v8f acc[4] = {};
  gemm_nb<4>(h1f, DPAD, wih1, DPAD, m0, n0, DPAD, acc);
  gemm_nb<4>(h2mf, DPAD, whh1, DPAD, m0, n0, DPAD, acc);
#pragma unroll
  for (int i = 0; i < 4; ++i) {
    int col = n0 + i * 16 + r15;
    float bias = bih1[col] + bhh1[col];
#pragma unroll
    for (int r = 0; r < 8; ++r)
      gates[(long)(m0 + half * 8 + r) * G4 + col] = acc[i][r] + bias;
  }
}

__global__ void k_cell1_elem(const float* __restrict__ gates, const int* __restrict__ term_t,
                             _Float16* __restrict__ h2f, float* __restrict__ c2,
                             float* __restrict__ outs_t) {
  int idx = blockIdx.x * blockDim.x + threadIdx.x;
  if (idx >= BB * DD) return;
  int b = idx / DD, j = idx % DD;
  const float* g = gates + (long)b * G4;
  float gi = g[j], gf = g[DD + j], gg = g[2 * DD + j], go = g[3 * DD + j];
  float nd = term_t[b] ? 0.0f : 1.0f;
  float cp = c2[b * DPAD + j] * nd;
  float cn = sigmoidf_(gf) * cp + sigmoidf_(gi) * tanhf(gg);
  float hn = sigmoidf_(go) * tanhf(cn);
  c2[b * DPAD + j] = cn;
  h2f[b * DPAD + j] = (_Float16)hn;
  outs_t[(long)b * DD + j] = hn;
}

// ---------------- head: policy logits, baseline, argmax ----------------
__global__ void k_head(const float* __restrict__ outs, const float* __restrict__ polw,
                       const float* __restrict__ polb, const float* __restrict__ basew,
                       const float* __restrict__ baseb, float* __restrict__ logits,
                       float* __restrict__ baseline, int* __restrict__ action) {
  int m = blockIdx.x * blockDim.x + threadIdx.x;
  if (m >= TB) return;
  const float* x = outs + (long)m * DD;
  float acc[AA];
#pragma unroll
  for (int a = 0; a < AA; ++a) acc[a] = polb[a];
  float bl = baseb[0];
  for (int j = 0; j < DD; ++j) {
    float xv = x[j];
#pragma unroll
    for (int a = 0; a < AA; ++a) acc[a] += xv * polw[a * DD + j];
    bl += xv * basew[j];
  }
  int best = 0;
  float bv = acc[0];
#pragma unroll
  for (int a = 0; a < AA; ++a) {
    logits[(long)m * AA + a] = acc[a];
    if (acc[a] > bv) { bv = acc[a]; best = a; }
  }
  baseline[m] = bl;
  action[m] = best;
}

// ---------------- host launcher ----------------
extern "C" void kernel_launch(void* const* d_in, const int* in_sizes, int n_in,
                              void* d_out, int out_size, void* d_ws, size_t ws_size,
                              hipStream_t stream) {
  const float* obs      = (const float*)d_in[0];
  const int*   last_act = (const int*)  d_in[1];
  const float* reward   = (const float*)d_in[2];
  const int*   term     = (const int*)  d_in[3];
  const float* enc_w    = (const float*)d_in[4];
  const float* enc_b    = (const float*)d_in[5];
  const float* w_ih0    = (const float*)d_in[6];
  const float* w_hh0    = (const float*)d_in[7];
  const float* b_ih0    = (const float*)d_in[8];
  const float* b_hh0    = (const float*)d_in[9];
  const float* w_ih1    = (const float*)d_in[10];
  const float* w_hh1    = (const float*)d_in[11];
  const float* b_ih1    = (const float*)d_in[12];
  const float* b_hh1    = (const float*)d_in[13];
  const float* pol_w    = (const float*)d_in[14];
  const float* pol_b    = (const float*)d_in[15];
  const float* base_w   = (const float*)d_in[16];
  const float* base_b   = (const float*)d_in[17];

  float* out_logits   = (float*)d_out;
  float* out_baseline = out_logits + (size_t)TB * AA;
  int*   out_action   = (int*)(out_baseline + TB);

  // workspace carve-up (256B aligned)
  char*  w   = (char*)d_ws;
  size_t off = 0;
  auto carve = [&](size_t bytes) -> void* {
    off = (off + 255) & ~(size_t)255;
    void* p = w + off;
    off += bytes;
    return p;
  };
  _Float16* obs_h   = (_Float16*)carve((size_t)TB * OBSN * 2);     // 64 MB
  _Float16* encw_h  = (_Float16*)carve((size_t)FEAT * OBSN * 2);
  _Float16* wih0_h  = (_Float16*)carve((size_t)G4 * DPAD * 2);
  _Float16* whh0_h  = (_Float16*)carve((size_t)G4 * DPAD * 2);
  _Float16* wih1_h  = (_Float16*)carve((size_t)G4 * DPAD * 2);
  _Float16* whh1_h  = (_Float16*)carve((size_t)G4 * DPAD * 2);
  _Float16* lstm_in = (_Float16*)carve((size_t)TB * DPAD * 2);     // ~36 MB
  _Float16* gates_x = (_Float16*)carve((size_t)TB * G4 * 2);       // ~138 MB
  float*    gatesb  = (float*)   carve((size_t)BB * G4 * 4);
  // contiguous state block: h1f,h2f,h1mf,h2mf (f16) + c1,c2 (f32); zeroed each call
  char*     stateblk = (char*)carve((size_t)4 * BB * DPAD * 2 + (size_t)2 * BB * DPAD * 4);
  float*    outs    = (float*)   carve((size_t)TB * DD * 4);       // ~69 MB
  _Float16* h1f  = (_Float16*)stateblk;
  _Float16* h2f  = h1f + (size_t)BB * DPAD;
  _Float16* h1mf = h2f + (size_t)BB * DPAD;
  _Float16* h2mf = h1mf + (size_t)BB * DPAD;
  float*    c1   = (float*)(h2mf + (size_t)BB * DPAD);
  float*    c2   = c1 + (size_t)BB * DPAD;

  // 1) convert obs + weights to f16 (weights padded K: 528 -> 544)
  k_cvt<<<(TB * OBSN) / 256, 256, 0, stream>>>(obs, obs_h, (long)TB * OBSN);
  {
    long n = (long)FEAT * OBSN;
    k_cvt_pad<<<(int)((n + 255) / 256), 256, 0, stream>>>(enc_w, encw_h, FEAT, OBSN, OBSN);
    n = (long)G4 * DPAD;
    int g = (int)((n + 255) / 256);
    k_cvt_pad<<<g, 256, 0, stream>>>(w_ih0, wih0_h, G4, DD, DPAD);
    k_cvt_pad<<<g, 256, 0, stream>>>(w_hh0, whh0_h, G4, DD, DPAD);
    k_cvt_pad<<<g, 256, 0, stream>>>(w_ih1, wih1_h, G4, DD, DPAD);
    k_cvt_pad<<<g, 256, 0, stream>>>(w_hh1, whh1_h, G4, DD, DPAD);
  }
  // 2) zero LSTM state block (h pads must stay zero for WMMA A loads)
  {
    long nbytes = (long)4 * BB * DPAD * 2 + (long)2 * BB * DPAD * 4;
    long n = nbytes / 4;
    k_zero<<<(int)((n + 255) / 256), 256, 0, stream>>>((float*)stateblk, n);
  }
  // 3) encoder GEMM + extras  (2048 Mtiles * 8 Ngroups = 16384 waves -> 4096 blocks exact)
  k_enc_gemm<<<(TB / 16) * (FEAT / 64) / 4, 128, 0, stream>>>(obs_h, encw_h, enc_b, lstm_in);
  k_fill_extras<<<TB / 256, 256, 0, stream>>>(reward, last_act, lstm_in);
  // 4) hoisted x-projection   (2048 * 33 = 67584 waves -> 16896 blocks exact)
  k_gatesx_gemm<<<(TB / 16) * (G4 / 64) / 4, 128, 0, stream>>>(lstm_in, wih0_h, b_ih0, b_hh0, gates_x);
  // 5) sequential scan: 5 stream-ordered kernels per timestep
  const int cell_gemm_blocks = (16 * (G4 / 64)) / 4;   // 132 exact
  const int elem_blocks      = (BB * DD) / 256;        // 528 exact
  const int prep_blocks      = (BB * DPAD) / 256;      // 544 exact
  for (int t = 0; t < TT; ++t) {
    const _Float16* gx_t   = gates_x + (size_t)t * BB * G4;
    const int*      term_t = term + (size_t)t * BB;
    float*          outs_t = outs + (size_t)t * BB * DD;
    k_prep_mask<<<prep_blocks, 256, 0, stream>>>(h1f, h2f, term_t, h1mf, h2mf);
    k_cell0_gemm<<<cell_gemm_blocks, 128, 0, stream>>>(h1mf, whh0_h, gx_t, gatesb);
    k_cell0_elem<<<elem_blocks, 256, 0, stream>>>(gatesb, term_t, h1f, c1);
    k_cell1_gemm<<<cell_gemm_blocks, 128, 0, stream>>>(h1f, h2mf, wih1_h, whh1_h, b_ih1, b_hh1, gatesb);
    k_cell1_elem<<<elem_blocks, 256, 0, stream>>>(gatesb, term_t, h2f, c2, outs_t);
  }
  // 6) policy / baseline / argmax
  k_head<<<TB / 256, 256, 0, stream>>>(outs, pol_w, pol_b, base_w, base_b,
                                       out_logits, out_baseline, out_action);
}
